// e2_cgrl_7241314861553
// MI455X (gfx1250) — matrix operand order repaired and checked by reading, not compiled, not run on hardware
//
#include <hip/hip_runtime.h>
#include <hip/hip_bf16.h>
#include <stdint.h>

// ---------------------------------------------------------------------------
// Problem constants (from reference)
// ---------------------------------------------------------------------------
constexpr int N_NODES = 50000;
constexpr int F_IN    = 500;
constexpr int K_PAD   = 512;   // F_IN padded to multiple of 32 for WMMA K
constexpr int H_DIM   = 256;
constexpr int N_EDGE  = 800000;
constexpr int R_VIEW  = 3;

constexpr uint32_t SEED_IN  = 0x9E3779B9u ^ 42u;
constexpr uint32_t SEED_HID = 0x85EBCA6Bu ^ 42u;
constexpr uint32_t KEEP_THRESH = 3435973837u;  // 0.8 * 2^32
constexpr float    DROP_SCALE  = 1.25f;        // 1 / 0.8

typedef __attribute__((ext_vector_type(16))) __bf16        v16bf;
typedef __attribute__((ext_vector_type(8)))  float         v8f;
typedef __attribute__((ext_vector_type(8)))  unsigned int  v8u;

__device__ __forceinline__ uint32_t hash32(uint32_t x, uint32_t seed) {
    x ^= seed;
    x *= 0x9E3779B9u; x ^= x >> 16;
    x *= 0x85EBCA6Bu; x ^= x >> 13;
    x *= 0xC2B2AE35u; x ^= x >> 16;
    return x;
}

// gfx1250 generic pointers to LDS carry the LDS byte offset in addr[31:0]
// (ISA 10.2 aperture rules), which is exactly the VDST operand format of
// GLOBAL_LOAD_ASYNC_TO_LDS (the LDS byte address is taken from that VGPR plus the offset).
__device__ __forceinline__ unsigned lds_off(const void* p) {
    return (unsigned)(uintptr_t)p;
}

// Async copy 16B from global to LDS (memory -> LDS, tracked by ASYNCcnt).
__device__ __forceinline__ void async_copy_b128(unsigned lds, const void* gaddr) {
    asm volatile("global_load_async_to_lds_b128 %0, %1, off"
                 :: "v"(lds), "v"(gaddr) : "memory");
}

__device__ __forceinline__ void wait_async_le5() {
    asm volatile("s_wait_asynccnt 0x5" ::: "memory");
}
__device__ __forceinline__ void wait_async_0() {
    asm volatile("s_wait_asynccnt 0x0" ::: "memory");
}

// ---------------------------------------------------------------------------
// 1) dropout(seq_a) -> bf16, padded [N, K_PAD]
// ---------------------------------------------------------------------------
__global__ __launch_bounds__(256) void prep_x_kernel(
    const float* __restrict__ seq_a, __bf16* __restrict__ x_bf)
{
    size_t idx = (size_t)blockIdx.x * 256 + threadIdx.x;
    if (idx >= (size_t)N_NODES * K_PAD) return;
    int n = (int)(idx / K_PAD);
    int k = (int)(idx % K_PAD);
    float v = 0.0f;
    if (k < F_IN) {
        v = seq_a[(size_t)n * F_IN + k];
        uint32_t h = hash32((uint32_t)(n * F_IN + k), SEED_IN);
        v = (h < KEEP_THRESH) ? v * DROP_SCALE : 0.0f;
    }
    x_bf[idx] = (__bf16)v;
}

// ---------------------------------------------------------------------------
// 2) W1 [F_IN,H] -> W1t bf16 [H, K_PAD];  W2 [H,H] -> W2t bf16 [H, H]
//    (transposed so LDS staging / B-fragment reads are contiguous)
// ---------------------------------------------------------------------------
__global__ __launch_bounds__(256) void prep_w_kernel(
    const float* __restrict__ W1, const float* __restrict__ W2,
    __bf16* __restrict__ W1t, __bf16* __restrict__ W2t)
{
    int idx = blockIdx.x * 256 + threadIdx.x;
    if (idx < K_PAD * H_DIM) {                       // W1t[n][k]
        int n = idx / K_PAD, k = idx % K_PAD;
        float v = (k < F_IN) ? W1[(size_t)k * H_DIM + n] : 0.0f;
        W1t[idx] = (__bf16)v;
    } else if (idx < K_PAD * H_DIM + H_DIM * H_DIM) { // W2t[n][k]
        int j = idx - K_PAD * H_DIM;
        int n = j / H_DIM, k = j % H_DIM;
        W2t[j] = (__bf16)W2[(size_t)k * H_DIM + n];
    }
}

// ---------------------------------------------------------------------------
// 3) Tiled bf16 WMMA GEMM:  out[M,256] = A[M,KDIM] @ Bt[256,KDIM]^T + bias
//    Block tile: 64 rows x 256 cols, 256 threads (8 waves).
//    Wave w: M-frag = w&3 (16 rows), N-half = w>>2 (8 x 16-col frags).
//    - Double-buffered LDS fed by GLOBAL_LOAD_ASYNC_TO_LDS_B128 (ASYNCcnt):
//      each wave issues exactly 5 async b128 per tile (1 A + 4 B chunks);
//      async loads complete in order, so s_wait_asynccnt 5 retires the
//      current tile while the next is in flight.
//    - B fragments software-pipelined in registers: frag j+1 is ds_loaded
//      before WMMA j issues, so DS latency hides under the XDL pipe.
//    LAYER1: relu -> bf16 out.   LAYER2: fp32 h_a + dropout -> h_p_0.
// ---------------------------------------------------------------------------
template <int KDIM, bool LAYER1>
__global__ __launch_bounds__(256) void gemm_wmma_kernel(
    const __bf16* __restrict__ A, const __bf16* __restrict__ Bt,
    const float* __restrict__ bias,
    __bf16* __restrict__ out_bf,     // LAYER1
    float*  __restrict__ out_f32,    // LAYER2: h_a
    float*  __restrict__ out_hp,     // LAYER2: dropout(h_a)
    int nrows)
{
    __shared__ __bf16 Alds[2][64 * 32];    // [buf][row][k]  2 x 4 KB
    __shared__ __bf16 Blds[2][256 * 32];   // [buf][n][k]    2 x 16 KB
    constexpr int NSTEP = KDIM / 32;

    const int t    = threadIdx.x;
    const int wave = t >> 5;
    const int lane = t & 31;
    const int mf   = wave & 3;          // which 16-row fragment
    const int nh   = wave >> 2;         // which 128-col half
    const int gm0  = blockIdx.x * 64;

    const int l15  = lane & 15;
    const int hi8  = (lane >= 16) ? 8 : 0;
    const int hi16 = (lane >= 16) ? 16 : 0;

    // staging assignment (branch-free: clamp OOB rows; their outputs are
    // never stored, and EXEC stays all-ones as WMMA requires)
    const int arow  = t >> 2;              // 0..63
    const int akoff = (t & 3) * 8;         // 0,8,16,24
    const int agr   = min(gm0 + arow, nrows - 1);
    const __bf16* agp = A  + (size_t)agr * KDIM + akoff;
    const __bf16* bgp = Bt + (size_t)t * KDIM;

    auto issue_tile = [&](int k0, int buf) {
        async_copy_b128(lds_off(&Alds[buf][arow * 32 + akoff]), agp + k0);
        #pragma unroll
        for (int j = 0; j < 4; ++j)
            async_copy_b128(lds_off(&Blds[buf][t * 32 + j * 8]), bgp + k0 + j * 8);
    };

    // B-fragment register load (2 runs of 4 consecutive u32 = 2 x ds_load_b128)
    auto load_bfrag = [&](v8u& bu, int buf, int j) {
        int n = nh * 128 + j * 16 + l15;
        #pragma unroll
        for (int p = 0; p < 8; ++p) {
            int kk = (p & 3) * 2 + (p >> 2) * 16 + hi8;   // K pair index
            bu[p] = *(const unsigned int*)(&Blds[buf][n * 32 + kk]);
        }
    };

    v8f acc[8] = {};

    issue_tile(0, 0);
    for (int s = 0; s < NSTEP; ++s) {
        const int buf = s & 1;
        if (s + 1 < NSTEP) {
            issue_tile((s + 1) * 32, buf ^ 1);  // prefetch next tile
            wait_async_le5();                   // current tile retired
        } else {
            wait_async_0();
        }
        __syncthreads();

        // ---- A fragment (16x32 bf16, ISA 7.12.2 layout) ----
        v8u au;
        const int ar = mf * 16 + l15;
        #pragma unroll
        for (int p = 0; p < 8; ++p) {
            int kk = (p & 3) * 2 + (p >> 2) * 16 + hi8;   // K pair index
            au[p] = *(const unsigned int*)(&Alds[buf][ar * 32 + kk]);
        }
        v16bf afrag = __builtin_bit_cast(v16bf, au);

        // ---- 8 B fragments, double-buffered in registers + WMMA ----
        v8u bu[2];
        load_bfrag(bu[0], buf, 0);
        #pragma unroll
        for (int j = 0; j < 8; ++j) {
            if (j + 1 < 8) load_bfrag(bu[(j + 1) & 1], buf, j + 1);
            v16bf bfrag = __builtin_bit_cast(v16bf, bu[j & 1]);
            acc[j] = __builtin_amdgcn_wmma_f32_16x16x32_bf16(
                false, afrag, false, bfrag, (short)0, acc[j], false, false);
        }
        __syncthreads();   // all waves done reading before buf is re-filled
    }

    // ---- epilogue ----
    #pragma unroll
    for (int j = 0; j < 8; ++j) {
        int gn = nh * 128 + j * 16 + l15;
        float bb = bias[gn];
        #pragma unroll
        for (int jv = 0; jv < 8; ++jv) {
            int m  = mf * 16 + jv + hi8;                   // C/D layout
            int gm = gm0 + m;
            if (gm < nrows) {
                float v = acc[j][jv] + bb;
                size_t o = (size_t)gm * H_DIM + gn;
                if (LAYER1) {
                    v = v > 0.0f ? v : 0.0f;
                    out_bf[o] = (__bf16)v;
                } else {
                    out_f32[o] = v;
                    uint32_t h = hash32((uint32_t)o, SEED_HID);
                    out_hp[o] = (h < KEEP_THRESH) ? v * DROP_SCALE : 0.0f;
                }
            }
        }
    }
}

// ---------------------------------------------------------------------------
// 4) zero the scatter target (stack region of d_out)
// ---------------------------------------------------------------------------
__global__ __launch_bounds__(256) void zero_kernel(float4* __restrict__ p, size_t n4)
{
    size_t i = (size_t)blockIdx.x * 256 + threadIdx.x;
    if (i < n4) p[i] = make_float4(0.f, 0.f, 0.f, 0.f);
}

// ---------------------------------------------------------------------------
// 5) SpMM scatter: one wave per edge, 8 columns/lane, f32 L2 atomics.
//    h_p_0 is 51.2 MB and the fp32 stack is 153.6 MB.
// ---------------------------------------------------------------------------
__global__ __launch_bounds__(256) void spmm_scatter_kernel(
    const int* __restrict__ rows, const int* __restrict__ cols,
    const float* __restrict__ vals, const float* __restrict__ hp0,
    float* __restrict__ out_stack)
{
    const int lane = threadIdx.x & 31;
    const long long ge = (long long)blockIdx.x * 8 + (threadIdx.x >> 5);
    if (ge >= (long long)R_VIEW * N_EDGE) return;
    const int r = (int)(ge / N_EDGE);
    const int e = (int)(ge % N_EDGE);
    const size_t eo = (size_t)r * N_EDGE + e;

    const int   row = rows[eo];
    const int   col = cols[eo];
    const float v   = vals[eo];

    const float4* src = (const float4*)(hp0 + (size_t)col * H_DIM);
    float4 d0 = src[lane * 2];
    float4 d1 = src[lane * 2 + 1];

    float* dst = out_stack + ((size_t)r * N_NODES + row) * H_DIM + lane * 8;
    __hip_atomic_fetch_add(dst + 0, v * d0.x, __ATOMIC_RELAXED, __HIP_MEMORY_SCOPE_AGENT);
    __hip_atomic_fetch_add(dst + 1, v * d0.y, __ATOMIC_RELAXED, __HIP_MEMORY_SCOPE_AGENT);
    __hip_atomic_fetch_add(dst + 2, v * d0.z, __ATOMIC_RELAXED, __HIP_MEMORY_SCOPE_AGENT);
    __hip_atomic_fetch_add(dst + 3, v * d0.w, __ATOMIC_RELAXED, __HIP_MEMORY_SCOPE_AGENT);
    __hip_atomic_fetch_add(dst + 4, v * d1.x, __ATOMIC_RELAXED, __HIP_MEMORY_SCOPE_AGENT);
    __hip_atomic_fetch_add(dst + 5, v * d1.y, __ATOMIC_RELAXED, __HIP_MEMORY_SCOPE_AGENT);
    __hip_atomic_fetch_add(dst + 6, v * d1.z, __ATOMIC_RELAXED, __HIP_MEMORY_SCOPE_AGENT);
    __hip_atomic_fetch_add(dst + 7, v * d1.w, __ATOMIC_RELAXED, __HIP_MEMORY_SCOPE_AGENT);
}

// ---------------------------------------------------------------------------
// 6) fusion mean over the 3 views
// ---------------------------------------------------------------------------
__global__ __launch_bounds__(256) void fuse_mean_kernel(
    const float* __restrict__ stack, float* __restrict__ out)
{
    size_t i = (size_t)blockIdx.x * 256 + threadIdx.x;
    constexpr size_t NH = (size_t)N_NODES * H_DIM;
    if (i >= NH) return;
    out[i] = (stack[i] + stack[NH + i] + stack[2 * NH + i]) * (1.0f / 3.0f);
}

// ---------------------------------------------------------------------------
// launch
// ---------------------------------------------------------------------------
extern "C" void kernel_launch(void* const* d_in, const int* in_sizes, int n_in,
                              void* d_out, int out_size, void* d_ws, size_t ws_size,
                              hipStream_t stream)
{
    const float* seq_a = (const float*)d_in[0];
    const float* vals  = (const float*)d_in[1];
    const float* W1    = (const float*)d_in[2];
    const float* b1    = (const float*)d_in[3];
    const float* W2    = (const float*)d_in[4];
    const float* b2    = (const float*)d_in[5];
    const int*   rows  = (const int*)d_in[6];
    const int*   cols  = (const int*)d_in[7];

    float* out_f = (float*)d_out;
    constexpr size_t NH = (size_t)N_NODES * H_DIM;
    float* out_ha     = out_f;                 // [N,H]
    float* out_stack  = out_f + NH;            // [R,N,H]
    float* out_fusion = out_f + NH + (size_t)R_VIEW * NH;

    // workspace layout (~128.4 MB)
    char* p = (char*)d_ws;
    __bf16* x_bf = (__bf16*)p;  p += (size_t)N_NODES * K_PAD * sizeof(__bf16); // 51.2 MB
    __bf16* h_bf = (__bf16*)p;  p += (size_t)N_NODES * H_DIM * sizeof(__bf16); // 25.6 MB
    float*  hp0  = (float*)p;   p += (size_t)N_NODES * H_DIM * sizeof(float);  // 51.2 MB
    __bf16* W1t  = (__bf16*)p;  p += (size_t)K_PAD * H_DIM * sizeof(__bf16);   // 256 KB
    __bf16* W2t  = (__bf16*)p;  p += (size_t)H_DIM * H_DIM * sizeof(__bf16);   // 128 KB
    (void)ws_size; (void)in_sizes; (void)n_in; (void)out_size;

    // 1) dropout + bf16 pack of seq_a
    {
        size_t tot = (size_t)N_NODES * K_PAD;
        prep_x_kernel<<<(unsigned)((tot + 255) / 256), 256, 0, stream>>>(seq_a, x_bf);
    }
    // 2) transpose + bf16 pack of weights
    {
        int tot = K_PAD * H_DIM + H_DIM * H_DIM;
        prep_w_kernel<<<(tot + 255) / 256, 256, 0, stream>>>(W1, W2, W1t, W2t);
    }
    // 3) h = relu(x @ W1 + b1)   (bf16 out)
    {
        int grid = (N_NODES + 63) / 64;
        gemm_wmma_kernel<K_PAD, true><<<grid, 256, 0, stream>>>(
            x_bf, W1t, b1, h_bf, nullptr, nullptr, N_NODES);
    }
    // 4) h_a = h @ W2 + b2 ; h_p_0 = dropout(h_a)
    {
        int grid = (N_NODES + 63) / 64;
        gemm_wmma_kernel<H_DIM, false><<<grid, 256, 0, stream>>>(
            h_bf, W2t, b2, nullptr, out_ha, hp0, N_NODES);
    }
    // 5) zero stack, scatter SpMM x3
    {
        size_t n4 = (size_t)R_VIEW * NH / 4;
        zero_kernel<<<(unsigned)((n4 + 255) / 256), 256, 0, stream>>>(
            (float4*)out_stack, n4);
        long long waves = (long long)R_VIEW * N_EDGE;
        spmm_scatter_kernel<<<(unsigned)((waves + 7) / 8), 256, 0, stream>>>(
            rows, cols, vals, hp0, out_stack);
    }
    // 6) fusion mean
    fuse_mean_kernel<<<(unsigned)((NH + 255) / 256), 256, 0, stream>>>(
        out_stack, out_fusion);
}
